// SimpleVQVAE_11759620457004
// MI455X (gfx1250) — matrix operand order repaired and checked by reading, not compiled, not run on hardware
//
#include <hip/hip_runtime.h>
#include <hip/hip_bf16.h>

// ---------------------------------------------------------------------------
// VQ-VAE forward for MI455X (gfx1250):
//  - bf16 WMMA (v_wmma_f32_16x16x32_bf16), fp32 accumulate
//  - async global->LDS tile staging (global_load_async_to_lds_b128 + ASYNCcnt)
//    with double-buffered software pipeline
//  - fused VQ argmin (never materializes the 32768x8192 distance matrix)
// ---------------------------------------------------------------------------

typedef __attribute__((ext_vector_type(16))) __bf16 v16bf;
typedef __attribute__((ext_vector_type(8)))  float  v8f;

union Frag {
    uint4 q[2];
    v16bf v;
};
static_assert(sizeof(v16bf) == 32, "v16bf must be 32B");

__device__ __forceinline__ unsigned short f2bf(float f) {
    unsigned int u = __float_as_uint(f);
    unsigned int r = u + 0x7FFFu + ((u >> 16) & 1u);   // round-to-nearest-even
    return (unsigned short)(r >> 16);
}
__device__ __forceinline__ unsigned int pk2(float a, float b) {
    return (unsigned int)f2bf(a) | ((unsigned int)f2bf(b) << 16);
}

// Async copy 16B global -> LDS (CDNA5, tracked by ASYNCcnt).
// lds_off = low 32 bits of the generic shared-memory address (= LDS byte offset).
__device__ __forceinline__ void async_b128(unsigned lds_off, const void* gptr) {
    asm volatile("global_load_async_to_lds_b128 %0, %1, off"
                 :: "v"(lds_off), "v"(gptr)
                 : "memory");
}
__device__ __forceinline__ unsigned lds_off32(const void* p) {
    return (unsigned)(unsigned long long)p;
}

// ---------------------------------------------------------------------------
// Prep kernels
// ---------------------------------------------------------------------------

// W[K][N] fp32 -> WT[N][K] bf16
__global__ __launch_bounds__(256) void wtrans_kernel(const float* __restrict__ W,
                                                     unsigned short* __restrict__ WT,
                                                     int K, int N) {
    int i = blockIdx.x * 256 + threadIdx.x;
    if (i >= K * N) return;
    int k = i / N, n = i % N;
    WT[(size_t)n * K + k] = f2bf(W[i]);
}

// emb[8192][256] fp32 -> bf16 copy + row sum of squares
__global__ __launch_bounds__(256) void emb_prep_kernel(const float* __restrict__ emb,
                                                       unsigned short* __restrict__ embb,
                                                       float* __restrict__ esq) {
    __shared__ float s[256];
    int r = blockIdx.x, c = threadIdx.x;
    float v = emb[(size_t)r * 256 + c];
    embb[(size_t)r * 256 + c] = f2bf(v);
    s[c] = v * v;
    __syncthreads();
    for (int st = 128; st > 0; st >>= 1) {
        if (c < st) s[c] += s[c + st];
        __syncthreads();
    }
    if (c == 0) esq[r] = s[0];
}

// ---------------------------------------------------------------------------
// Tiled GEMM: out = act(A[M,K] @ W[K,N] + bias), W given transposed bf16.
// Block tile 128x128, 256 threads = 8 waves, wave tile 32(M) x 64(N),
// 2x4 WMMA 16x16 tiles per wave, K chunked by 32, LDS double-buffered with
// async global->LDS prefetch of the next K-chunk.
// ---------------------------------------------------------------------------
template <bool AF32>
__global__ __launch_bounds__(256) void gemm_kernel(const void* __restrict__ Ap,
                                                   const unsigned short* __restrict__ WT,
                                                   const float* __restrict__ bias,
                                                   float* __restrict__ outF,
                                                   unsigned short* __restrict__ outB,
                                                   int M, int N, int K, int relu) {
    __shared__ __align__(16) unsigned short As[2][128 * 40];   // [buf][row*40 + k]
    __shared__ __align__(16) unsigned short Bs[2][128 * 40];

    const int tid  = threadIdx.x;
    const int lane = tid & 31;
    const int wave = tid >> 5;
    const int wm   = wave & 3;          // 4 M-groups of 32
    const int wn   = wave >> 2;         // 2 N-groups of 64
    const int bm   = blockIdx.y * 128;
    const int bn   = blockIdx.x * 128;
    const int r16  = lane & 15;
    const int hi   = lane >> 4;
    const int srow = tid >> 1;          // staging: 128 rows x 2 segs of 16 elems
    const int sseg = tid & 1;

    constexpr int PER_TILE = AF32 ? 2 : 4;   // async loads issued per thread per tile

    auto stage = [&](int k0, int buf) {
        // B tile always async (WT is bf16 [N][K])
        {
            const unsigned short* g = WT + (size_t)(bn + srow) * K + k0 + sseg * 16;
            unsigned l = lds_off32(&Bs[buf][srow * 40 + sseg * 16]);
            async_b128(l, g);
            async_b128(l + 16, g + 8);
        }
        if (AF32) {
            const float* A = (const float*)Ap;
            const float4* src = (const float4*)(A + (size_t)(bm + srow) * K + k0 + sseg * 16);
            float4 f0 = src[0], f1 = src[1], f2 = src[2], f3 = src[3];
            unsigned int* dst = (unsigned int*)&As[buf][srow * 40 + sseg * 16];
            dst[0] = pk2(f0.x, f0.y); dst[1] = pk2(f0.z, f0.w);
            dst[2] = pk2(f1.x, f1.y); dst[3] = pk2(f1.z, f1.w);
            dst[4] = pk2(f2.x, f2.y); dst[5] = pk2(f2.z, f2.w);
            dst[6] = pk2(f3.x, f3.y); dst[7] = pk2(f3.z, f3.w);
        } else {
            const unsigned short* A = (const unsigned short*)Ap;
            const unsigned short* g = A + (size_t)(bm + srow) * K + k0 + sseg * 16;
            unsigned l = lds_off32(&As[buf][srow * 40 + sseg * 16]);
            async_b128(l, g);
            async_b128(l + 16, g + 8);
        }
    };

    v8f acc[2][4] = {};
    const int T = K >> 5;

    stage(0, 0);
    for (int t = 0; t < T; ++t) {
        const int cur = t & 1;
        if (t + 1 < T) {
            stage((t + 1) << 5, cur ^ 1);
            asm volatile("s_wait_asynccnt %0" :: "i"(PER_TILE) : "memory");
        } else {
            asm volatile("s_wait_asynccnt 0" ::: "memory");
        }
        __syncthreads();   // tile t resident for all waves

        Frag a[2], b[4];
#pragma unroll
        for (int mt = 0; mt < 2; ++mt) {
            int row = wm * 32 + mt * 16 + r16;
            int kb  = hi * 8;   // A layout: half 0 -> K 0-7 & 16-23; half 1 -> 8-15 & 24-31
            a[mt].q[0] = *(const uint4*)&As[cur][row * 40 + kb];
            a[mt].q[1] = *(const uint4*)&As[cur][row * 40 + kb + 16];
        }
#pragma unroll
        for (int nt = 0; nt < 4; ++nt) {
            int col = wn * 64 + nt * 16 + r16;
            int kb  = hi * 8;   // B layout: half 0 -> K 0-7 & 16-23; half 1 -> 8-15 & 24-31
            b[nt].q[0] = *(const uint4*)&Bs[cur][col * 40 + kb];
            b[nt].q[1] = *(const uint4*)&Bs[cur][col * 40 + kb + 16];
        }

#pragma unroll
        for (int mt = 0; mt < 2; ++mt)
#pragma unroll
            for (int nt = 0; nt < 4; ++nt)
                acc[mt][nt] = __builtin_amdgcn_wmma_f32_16x16x32_bf16(
                    false, a[mt].v, false, b[nt].v, (short)0, acc[mt][nt], false, false);

        __syncthreads();   // reads done before this buffer is overwritten
    }

    // ---- epilogue: bias + (relu) + stores ----
#pragma unroll
    for (int mt = 0; mt < 2; ++mt) {
#pragma unroll
        for (int nt = 0; nt < 4; ++nt) {
            int mbase = bm + wm * 32 + mt * 16 + hi * 8;
            int n     = bn + wn * 64 + nt * 16 + r16;
            float bv  = bias[n];
#pragma unroll
            for (int j = 0; j < 8; ++j) {
                float v = acc[mt][nt][j] + bv;
                if (relu) v = fmaxf(v, 0.0f);
                size_t off = (size_t)(mbase + j) * N + n;
                if (outF) outF[off] = v;
                if (outB) outB[off] = f2bf(v);
            }
        }
    }
}

// ---------------------------------------------------------------------------
// Fused VQ: indices[m] = argmin_n ( e_sq[n] - 2 * dot(z_e[m], emb[n]) )
// A = z_e in bf16 (zb), B = codebook bf16 [8192][256]; both staged async.
// ---------------------------------------------------------------------------
__global__ __launch_bounds__(256) void vq_kernel(const unsigned short* __restrict__ zb,
                                                 const unsigned short* __restrict__ embb,
                                                 const float* __restrict__ esq,
                                                 float* __restrict__ idxF,
                                                 int* __restrict__ idxI) {
    __shared__ __align__(16) unsigned short As[2][128 * 40];
    __shared__ __align__(16) unsigned short Bs[2][128 * 40];
    __shared__ float rv[128 * 2];
    __shared__ int   ri[128 * 2];

    const int tid  = threadIdx.x;
    const int lane = tid & 31;
    const int wave = tid >> 5;
    const int wm   = wave & 3;
    const int wn   = wave >> 2;
    const int bm   = blockIdx.x * 128;
    const int r16  = lane & 15;
    const int hi   = lane >> 4;
    const int srow = tid >> 1;
    const int sseg = tid & 1;
    const int K = 256;

    float bestv[2][8];
    int   besti[2][8];
#pragma unroll
    for (int mt = 0; mt < 2; ++mt)
#pragma unroll
        for (int j = 0; j < 8; ++j) { bestv[mt][j] = 3.4e38f; besti[mt][j] = 0; }

    auto stage = [&](int n0, int k0, int buf) {
        {
            const unsigned short* g = zb + (size_t)(bm + srow) * K + k0 + sseg * 16;
            unsigned l = lds_off32(&As[buf][srow * 40 + sseg * 16]);
            async_b128(l, g);
            async_b128(l + 16, g + 8);
        }
        {
            const unsigned short* g = embb + (size_t)(n0 + srow) * K + k0 + sseg * 16;
            unsigned l = lds_off32(&Bs[buf][srow * 40 + sseg * 16]);
            async_b128(l, g);
            async_b128(l + 16, g + 8);
        }
    };

    for (int n0 = 0; n0 < 8192; n0 += 128) {
        v8f acc[2][4] = {};
        const int T = K >> 5;   // 8
        stage(n0, 0, 0);
        for (int t = 0; t < T; ++t) {
            const int cur = t & 1;
            if (t + 1 < T) {
                stage(n0, (t + 1) << 5, cur ^ 1);
                asm volatile("s_wait_asynccnt 4" ::: "memory");
            } else {
                asm volatile("s_wait_asynccnt 0" ::: "memory");
            }
            __syncthreads();

            Frag a[2], b[4];
#pragma unroll
            for (int mt = 0; mt < 2; ++mt) {
                int row = wm * 32 + mt * 16 + r16;
                int kb  = hi * 8;
                a[mt].q[0] = *(const uint4*)&As[cur][row * 40 + kb];
                a[mt].q[1] = *(const uint4*)&As[cur][row * 40 + kb + 16];
            }
#pragma unroll
            for (int nt = 0; nt < 4; ++nt) {
                int col = wn * 64 + nt * 16 + r16;
                int kb  = hi * 8;
                b[nt].q[0] = *(const uint4*)&Bs[cur][col * 40 + kb];
                b[nt].q[1] = *(const uint4*)&Bs[cur][col * 40 + kb + 16];
            }
#pragma unroll
            for (int mt = 0; mt < 2; ++mt)
#pragma unroll
                for (int nt = 0; nt < 4; ++nt)
                    acc[mt][nt] = __builtin_amdgcn_wmma_f32_16x16x32_bf16(
                        false, a[mt].v, false, b[nt].v, (short)0, acc[mt][nt], false, false);
            __syncthreads();
        }

        // scores + running min
#pragma unroll
        for (int mt = 0; mt < 2; ++mt) {
#pragma unroll
            for (int nt = 0; nt < 4; ++nt) {
                int n = n0 + wn * 64 + nt * 16 + r16;
                float es = esq[n];
#pragma unroll
                for (int j = 0; j < 8; ++j) {
                    float s = es - 2.0f * acc[mt][nt][j];
                    if (s < bestv[mt][j]) { bestv[mt][j] = s; besti[mt][j] = n; }
                }
            }
        }
    }

    // cross-lane reduction over the 16 lanes holding the same row
#pragma unroll
    for (int mt = 0; mt < 2; ++mt) {
#pragma unroll
        for (int j = 0; j < 8; ++j) {
            float bv = bestv[mt][j];
            int   bi = besti[mt][j];
#pragma unroll
            for (int off = 8; off > 0; off >>= 1) {
                float ov = __shfl_xor(bv, off, 32);
                int   oi = __shfl_xor(bi, off, 32);
                if (ov < bv || (ov == bv && oi < bi)) { bv = ov; bi = oi; }
            }
            if (r16 == 0) {
                int rowlocal = wm * 32 + mt * 16 + j + hi * 8;
                rv[rowlocal * 2 + wn] = bv;
                ri[rowlocal * 2 + wn] = bi;
            }
        }
    }
    __syncthreads();

    if (tid < 128) {
        float v0 = rv[tid * 2 + 0], v1 = rv[tid * 2 + 1];
        int   i0 = ri[tid * 2 + 0], i1 = ri[tid * 2 + 1];
        int best = (v1 < v0 || (v1 == v0 && i1 < i0)) ? i1 : i0;
        int m = bm + tid;
        idxF[m] = (float)best;
        idxI[m] = best;
    }
}

// z_q gather: fp32 to output, bf16 to workspace for the decoder
__global__ __launch_bounds__(256) void gather_kernel(const float* __restrict__ emb,
                                                     const int* __restrict__ idxI,
                                                     float* __restrict__ zq,
                                                     unsigned short* __restrict__ zqb) {
    int i = blockIdx.x;
    int c = threadIdx.x;
    int id = idxI[i];
    float v = emb[(size_t)id * 256 + c];
    zq[(size_t)i * 256 + c]  = v;
    zqb[(size_t)i * 256 + c] = f2bf(v);
}

// ---------------------------------------------------------------------------
// Host launcher
// ---------------------------------------------------------------------------
extern "C" void kernel_launch(void* const* d_in, const int* in_sizes, int n_in,
                              void* d_out, int out_size, void* d_ws, size_t ws_size,
                              hipStream_t stream) {
    (void)in_sizes; (void)n_in; (void)out_size; (void)ws_size;

    const int N = 32768;

    const float* x   = (const float*)d_in[0];
    const float* W1  = (const float*)d_in[1];
    const float* b1  = (const float*)d_in[2];
    const float* W2  = (const float*)d_in[3];
    const float* b2  = (const float*)d_in[4];
    const float* W3  = (const float*)d_in[5];
    const float* b3  = (const float*)d_in[6];
    const float* emb = (const float*)d_in[7];
    const float* D1  = (const float*)d_in[8];
    const float* c1  = (const float*)d_in[9];
    const float* D2  = (const float*)d_in[10];
    const float* c2  = (const float*)d_in[11];
    const float* D3  = (const float*)d_in[12];
    const float* c3  = (const float*)d_in[13];

    float* out      = (float*)d_out;
    float* x_recon  = out;                                       // [32768,1024]
    float* z_e      = out + (size_t)N * 1024;                    // [32768,256]
    float* z_q      = z_e + (size_t)N * 256;                     // [32768,256]
    float* idxF     = z_q + (size_t)N * 256;                     // [32768]

    // workspace carve-up (256B aligned)
    char* ws = (char*)d_ws;
    size_t off = 0;
    auto carve = [&](size_t bytes) -> char* {
        off = (off + 255) & ~(size_t)255;
        char* p = ws + off;
        off += bytes;
        return p;
    };
    unsigned short* wt1  = (unsigned short*)carve((size_t)1024 * 512 * 2);
    unsigned short* wt2  = (unsigned short*)carve((size_t)512 * 256 * 2);
    unsigned short* wt3  = (unsigned short*)carve((size_t)256 * 256 * 2);
    unsigned short* dt1  = (unsigned short*)carve((size_t)256 * 256 * 2);
    unsigned short* dt2  = (unsigned short*)carve((size_t)256 * 512 * 2);
    unsigned short* dt3  = (unsigned short*)carve((size_t)512 * 1024 * 2);
    unsigned short* embb = (unsigned short*)carve((size_t)8192 * 256 * 2);
    float*          esq  = (float*)carve((size_t)8192 * 4);
    unsigned short* h1   = (unsigned short*)carve((size_t)N * 512 * 2);
    unsigned short* h2   = (unsigned short*)carve((size_t)N * 256 * 2);
    unsigned short* zb   = (unsigned short*)carve((size_t)N * 256 * 2);
    unsigned short* zqb  = (unsigned short*)carve((size_t)N * 256 * 2);
    unsigned short* g1   = (unsigned short*)carve((size_t)N * 256 * 2);
    unsigned short* g2   = (unsigned short*)carve((size_t)N * 512 * 2);
    int*            idxI = (int*)carve((size_t)N * 4);

    dim3 blk(256);

    // ---- weight prep ----
    wtrans_kernel<<<(1024 * 512 + 255) / 256, blk, 0, stream>>>(W1, wt1, 1024, 512);
    wtrans_kernel<<<(512 * 256 + 255) / 256, blk, 0, stream>>>(W2, wt2, 512, 256);
    wtrans_kernel<<<(256 * 256 + 255) / 256, blk, 0, stream>>>(W3, wt3, 256, 256);
    wtrans_kernel<<<(256 * 256 + 255) / 256, blk, 0, stream>>>(D1, dt1, 256, 256);
    wtrans_kernel<<<(256 * 512 + 255) / 256, blk, 0, stream>>>(D2, dt2, 256, 512);
    wtrans_kernel<<<(512 * 1024 + 255) / 256, blk, 0, stream>>>(D3, dt3, 512, 1024);
    emb_prep_kernel<<<8192, blk, 0, stream>>>(emb, embb, esq);

    // ---- encoder ----
    gemm_kernel<true><<<dim3(512 / 128, N / 128), blk, 0, stream>>>(
        x, wt1, b1, nullptr, h1, N, 512, 1024, 1);
    gemm_kernel<false><<<dim3(256 / 128, N / 128), blk, 0, stream>>>(
        h1, wt2, b2, nullptr, h2, N, 256, 512, 1);
    gemm_kernel<false><<<dim3(256 / 128, N / 128), blk, 0, stream>>>(
        h2, wt3, b3, z_e, zb, N, 256, 256, 0);

    // ---- vector quantize (fused argmin over 8192 codes) ----
    vq_kernel<<<N / 128, blk, 0, stream>>>(zb, embb, esq, idxF, idxI);
    gather_kernel<<<N, blk, 0, stream>>>(emb, idxI, z_q, zqb);

    // ---- decoder ----
    gemm_kernel<false><<<dim3(256 / 128, N / 128), blk, 0, stream>>>(
        zqb, dt1, c1, nullptr, g1, N, 256, 256, 1);
    gemm_kernel<false><<<dim3(512 / 128, N / 128), blk, 0, stream>>>(
        g1, dt2, c2, nullptr, g2, N, 512, 256, 1);
    gemm_kernel<false><<<dim3(1024 / 128, N / 128), blk, 0, stream>>>(
        g2, dt3, c3, x_recon, nullptr, N, 1024, 512, 0);
}
